// MambaBlock_56478819942752
// MI455X (gfx1250) — matrix operand order repaired and checked by reading, not compiled, not run on hardware
//
#include <hip/hip_runtime.h>
#include <hip/hip_bf16.h>

// ---------------------------------------------------------------------------
// Mamba block for MI455X (gfx1250, wave32, WMMA).
// GEMMs: bf16 v_wmma_f32_16x16x32_bf16 with LDS double-buffering fed by
// global_load_async_to_lds_b128 (ASYNCcnt pipelined). Scan: fp32 wave-parallel.
// ---------------------------------------------------------------------------

typedef __bf16 bf16_t;
typedef __attribute__((ext_vector_type(16))) __bf16 v16bf;
typedef __attribute__((ext_vector_type(8)))  float  v8f;

static constexpr int D_MODEL = 1024;
static constexpr int D_STATE = 16;
static constexpr int D_INNER = 2048;
static constexpr int DT_RANK = 64;
static constexpr int NBATCH  = 2;
static constexpr int SEQ     = 2048;
static constexpr int BL      = NBATCH * SEQ;          // 4096 token rows
static constexpr int XP_COLS = DT_RANK + 2 * D_STATE; // 96

// ---- workspace layout (bytes) ----
static constexpr size_t OFF_HS_BF   = 0;                       // 4096x1024 bf16
static constexpr size_t OFF_WIN_BF  = 8388608;                 // 4096x1024 bf16
static constexpr size_t OFF_WXP_BF  = 16777216;                // 96x2048  bf16
static constexpr size_t OFF_WDT_BF  = 17170432;                // 2048x64  bf16
static constexpr size_t OFF_WOUT_BF = 17432576;                // 1024x2048 bf16
static constexpr size_t OFF_X       = 21626880;                // 4096x2048 f32
static constexpr size_t OFF_Z       = 55181312;                // 4096x2048 f32
static constexpr size_t OFF_XC      = 88735744;                // 4096x2048 f32
static constexpr size_t OFF_XCBF    = 122290176;               // 4096x2048 bf16
static constexpr size_t OFF_XP      = 139067392;               // 4096x96  f32
static constexpr size_t OFF_XPBF    = 140640256;               // 4096x96  bf16
static constexpr size_t OFF_DT      = 141426688;               // 4096x2048 f32
static constexpr size_t OFF_YBF     = 174981120;               // 4096x2048 bf16

// ---------------------------------------------------------------------------
// CDNA5 async copy + counter helpers (inline asm; portable across toolchains)
// ---------------------------------------------------------------------------
template<int N>
__device__ __forceinline__ void wait_asynccnt() {
    asm volatile("s_wait_asynccnt %0" :: "i"(N) : "memory");
}

__device__ __forceinline__ void async_ld_b128(unsigned lds_off, const void* gaddr) {
    // LDS dest byte-offset in one VGPR, 64-bit global address in a VGPR pair.
    asm volatile("global_load_async_to_lds_b128 %0, %1, off"
                 :: "v"(lds_off), "v"((unsigned long long)gaddr) : "memory");
}

// ---------------------------------------------------------------------------
// fp32 -> bf16 cast (grid-stride)
// ---------------------------------------------------------------------------
__global__ void k_cast_bf16(const float* __restrict__ src, bf16_t* __restrict__ dst, int n) {
    int i = blockIdx.x * blockDim.x + threadIdx.x;
    int stride = gridDim.x * blockDim.x;
    for (; i < n; i += stride) dst[i] = (bf16_t)src[i];
}

// ---------------------------------------------------------------------------
// WMMA fragment loaders from LDS, per ISA 7.12.2 (wave32).
// A tile 16x32 (MxK): lanes 0-15 row M=lane, halves 0..7 = K[k0..k0+7],
//                     halves 8..15 = K[k0+16..k0+23]; lanes 16-31: +8 on K base.
// B tile 32x16 (KxN): lanes 0-15 col N=lane, same K order as A:
//                     halves 0..7 = K[k0..k0+7], 8..15 = K[k0+16..k0+23]; lanes 16-31: +8 on K base.
// ---------------------------------------------------------------------------
union FragU { v16bf v; uint4 q[2]; };

__device__ __forceinline__ v16bf frag_a(const bf16_t* __restrict__ p0, int ldp,
                                        int row, int kk, int lane) {
    FragU f;
    const bf16_t* p = p0 + row * ldp + kk + ((lane >> 4) << 3);
    f.q[0] = *(const uint4*)(p);        // K +0..+7
    f.q[1] = *(const uint4*)(p + 16);   // K +16..+23
    return f.v;
}

__device__ __forceinline__ v16bf frag_b(const bf16_t* __restrict__ p0, int ldp,
                                        int row, int kk, int lane) {
    FragU f;
    const bf16_t* p = p0 + row * ldp + kk + ((lane >> 4) << 3);
    f.q[0] = *(const uint4*)(p);        // K +0..+7
    f.q[1] = *(const uint4*)(p + 16);   // K +16..+23
    return f.v;
}

// ---------------------------------------------------------------------------
// LDS double-buffered WMMA GEMM: out[M,N] = A[M,K] * W[N,K]^T  (bf16 -> f32)
// MT/NT: 16x16 tiles per wave; WM/WN: wave grid per block (8 waves, 256 thr).
// BK: K per stage (one async-staged LDS tile). EPI: 0 plain, 1 x/z split,
// 2 bias+softplus (dt), 3 dual f32+bf16 store.
// ---------------------------------------------------------------------------
template<int MT, int NT, int WM, int WN, int BK, int EPI>
__global__ __launch_bounds__(256) void k_wmma_gemm(
        const bf16_t* __restrict__ A, int lda,
        const bf16_t* __restrict__ W, int ldw, int K,
        float* __restrict__ out0, float* __restrict__ out1,
        int ldc, int nsplit,
        const float* __restrict__ bias,
        bf16_t* __restrict__ outbf) {
    constexpr int BM  = MT * 16 * WM;
    constexpr int BN  = NT * 16 * WN;
    constexpr int LDP = BK + 8;              // +8 bf16 row pad vs LDS banks
    constexpr int CA  = BM * BK / 8;         // 16B chunks in A stage
    constexpr int CB  = BN * BK / 8;         // 16B chunks in B stage
    constexpr int PER = (CA + CB) / 256;     // async issues per thread/stage

    __shared__ bf16_t sA[2][BM * LDP];
    __shared__ bf16_t sB[2][BN * LDP];

    const int tid  = threadIdx.x;
    const int lane = tid & 31;
    const int wave = tid >> 5;
    const int wm = wave % WM;
    const int wn = wave / WM;
    const int rowA0 = blockIdx.x * BM;
    const int rowB0 = blockIdx.y * BN;

    auto issue = [&](int buf, int k0) {
        for (int c = tid; c < CA; c += 256) {
            const int r = c / (BK / 8), cc = c % (BK / 8);
            async_ld_b128((unsigned)(size_t)&sA[buf][r * LDP + cc * 8],
                          A + (size_t)(rowA0 + r) * lda + k0 + cc * 8);
        }
        for (int c = tid; c < CB; c += 256) {
            const int r = c / (BK / 8), cc = c % (BK / 8);
            async_ld_b128((unsigned)(size_t)&sB[buf][r * LDP + cc * 8],
                          W + (size_t)(rowB0 + r) * ldw + k0 + cc * 8);
        }
    };

    v8f acc[MT][NT];
    const v8f vzero = {0.f, 0.f, 0.f, 0.f, 0.f, 0.f, 0.f, 0.f};
#pragma unroll
    for (int i = 0; i < MT; ++i)
#pragma unroll
        for (int j = 0; j < NT; ++j) acc[i][j] = vzero;

    const int S = K / BK;
    issue(0, 0);

    const int arow = lane & 15;
    for (int s = 0; s < S; ++s) {
        if (s + 1 < S) {
            issue((s + 1) & 1, (s + 1) * BK);
            wait_asynccnt<PER>();    // stage s complete; stage s+1 in flight
        } else {
            wait_asynccnt<0>();
        }
        __syncthreads();
        const bf16_t* pa = &sA[s & 1][0];
        const bf16_t* pb = &sB[s & 1][0];
#pragma unroll
        for (int kk = 0; kk < BK; kk += 32) {
            v16bf af[MT], bfr[NT];
#pragma unroll
            for (int i = 0; i < MT; ++i)
                af[i] = frag_a(pa, LDP, wm * MT * 16 + i * 16 + arow, kk, lane);
#pragma unroll
            for (int j = 0; j < NT; ++j)
                bfr[j] = frag_b(pb, LDP, wn * NT * 16 + j * 16 + arow, kk, lane);
#pragma unroll
            for (int i = 0; i < MT; ++i)
#pragma unroll
                for (int j = 0; j < NT; ++j)
                    acc[i][j] = __builtin_amdgcn_wmma_f32_16x16x32_bf16(
                        false, af[i], false, bfr[j], (short)0, acc[i][j], false, false);
        }
        __syncthreads();             // stage buffer reusable after this
    }

    // Epilogue: C/D layout — VGPR r: lanes 0-15 -> M=r, lanes 16-31 -> M=r+8.
    const int mofs  = (lane >> 4) << 3;
    const int col16 = lane & 15;
#pragma unroll
    for (int i = 0; i < MT; ++i) {
#pragma unroll
        for (int j = 0; j < NT; ++j) {
#pragma unroll
            for (int r = 0; r < 8; ++r) {
                const int row = rowA0 + wm * MT * 16 + i * 16 + r + mofs;
                const int col = rowB0 + wn * NT * 16 + j * 16 + col16;
                float v = acc[i][j][r];
                if constexpr (EPI == 0) {
                    out0[(size_t)row * ldc + col] = v;
                } else if constexpr (EPI == 1) {
                    if (col < nsplit) out0[(size_t)row * ldc + col] = v;
                    else              out1[(size_t)row * ldc + (col - nsplit)] = v;
                } else if constexpr (EPI == 2) {
                    v += bias[col];
                    v = (v > 20.f) ? v : log1pf(__expf(v));   // softplus
                    out0[(size_t)row * ldc + col] = v;
                } else {
                    out0[(size_t)row * ldc + col] = v;
                    outbf[(size_t)row * ldc + col] = (bf16_t)v;
                }
            }
        }
    }
}

// ---------------------------------------------------------------------------
// Causal depthwise conv (K=4) + SiLU; writes f32 and bf16 copies.
// ---------------------------------------------------------------------------
__global__ void k_conv_silu(const float* __restrict__ x,
                            const float* __restrict__ cw,
                            const float* __restrict__ cb,
                            float* __restrict__ xc, bf16_t* __restrict__ xcbf) {
    long idx = (long)blockIdx.x * blockDim.x + threadIdx.x;
    if (idx >= (long)BL * D_INNER) return;
    const int c = (int)(idx & (D_INNER - 1));
    const long bl = idx >> 11;               // /2048
    const int l = (int)(bl & (SEQ - 1));
    float acc = cb[c];
#pragma unroll
    for (int k = 0; k < 4; ++k) {
        const int ls = l + k - 3;            // left pad K-1
        if (ls >= 0) acc += cw[c * 4 + k] * x[(bl + (k - 3)) * D_INNER + c];
    }
    const float sv = acc / (1.f + __expf(-acc));  // SiLU
    xc[idx] = sv;
    xcbf[idx] = (bf16_t)sv;
}

// ---------------------------------------------------------------------------
// Selective scan: lane = (channel pair, state). 16 lanes per channel reduce y
// over d_state via shfl_xor. Fuses y = (y + u*D) * silu(z) -> bf16 for out GEMM.
// block = 256 threads = 16 channels; grid = (D_INNER/16, NBATCH).
// ---------------------------------------------------------------------------
__global__ void k_scan(const float* __restrict__ dt, const float* __restrict__ u,
                       const float* __restrict__ xp, const float* __restrict__ z,
                       const float* __restrict__ A_log, const float* __restrict__ Dp,
                       bf16_t* __restrict__ ybf) {
    const int b = blockIdx.y;
    const int s = threadIdx.x & 15;
    const int d = blockIdx.x * 16 + (threadIdx.x >> 4);
    const float A  = -__expf(A_log[d * D_STATE + s]);
    const float Dv = Dp[d];
    float h = 0.f;
    const size_t base = (size_t)b * SEQ;
    for (int t = 0; t < SEQ; ++t) {
        const size_t r = base + t;
        if (t + 16 < SEQ) {                       // stream ahead (global_prefetch)
            __builtin_prefetch(&dt[(r + 16) * D_INNER + d], 0, 0);
            __builtin_prefetch(&u [(r + 16) * D_INNER + d], 0, 0);
        }
        const float dtv = dt[r * D_INNER + d];
        const float uv  = u [r * D_INNER + d];
        const float Bv  = xp[r * XP_COLS + DT_RANK + s];
        const float Cv  = xp[r * XP_COLS + DT_RANK + D_STATE + s];
        h = __expf(dtv * A) * h + (dtv * Bv) * uv;
        float p = h * Cv;
        p += __shfl_xor(p, 1, 16);
        p += __shfl_xor(p, 2, 16);
        p += __shfl_xor(p, 4, 16);
        p += __shfl_xor(p, 8, 16);
        if (s == 0) {
            const float zv = z[r * D_INNER + d];
            const float yv = (p + uv * Dv) * (zv / (1.f + __expf(-zv)));
            ybf[r * D_INNER + d] = (bf16_t)yv;
        }
    }
}

// ---------------------------------------------------------------------------
extern "C" void kernel_launch(void* const* d_in, const int* in_sizes, int n_in,
                              void* d_out, int out_size, void* d_ws, size_t ws_size,
                              hipStream_t stream) {
    const float* hs        = (const float*)d_in[0];
    const float* in_proj_w = (const float*)d_in[1];
    const float* conv_w    = (const float*)d_in[2];
    const float* conv_b    = (const float*)d_in[3];
    const float* x_proj_w  = (const float*)d_in[4];
    const float* dt_proj_w = (const float*)d_in[5];
    const float* dt_proj_b = (const float*)d_in[6];
    const float* A_log     = (const float*)d_in[7];
    const float* Dvec      = (const float*)d_in[8];
    const float* out_proj_w= (const float*)d_in[9];
    float* out = (float*)d_out;

    char* w = (char*)d_ws;
    bf16_t* hs_bf   = (bf16_t*)(w + OFF_HS_BF);
    bf16_t* win_bf  = (bf16_t*)(w + OFF_WIN_BF);
    bf16_t* wxp_bf  = (bf16_t*)(w + OFF_WXP_BF);
    bf16_t* wdt_bf  = (bf16_t*)(w + OFF_WDT_BF);
    bf16_t* wout_bf = (bf16_t*)(w + OFF_WOUT_BF);
    float*  xf      = (float*) (w + OFF_X);
    float*  zf      = (float*) (w + OFF_Z);
    float*  xcf     = (float*) (w + OFF_XC);
    bf16_t* xc_bf   = (bf16_t*)(w + OFF_XCBF);
    float*  xpf     = (float*) (w + OFF_XP);
    bf16_t* xp_bf   = (bf16_t*)(w + OFF_XPBF);
    float*  dtf     = (float*) (w + OFF_DT);
    bf16_t* y_bf    = (bf16_t*)(w + OFF_YBF);

    const int TPB = 256;
    auto gs = [](int n, int t) { return (n + t - 1) / t; };

    // 1) bf16 casts of activations + weights
    k_cast_bf16<<<gs(BL * D_MODEL, TPB), TPB, 0, stream>>>(hs, hs_bf, BL * D_MODEL);
    k_cast_bf16<<<gs(2 * D_INNER * D_MODEL, TPB), TPB, 0, stream>>>(in_proj_w, win_bf, 2 * D_INNER * D_MODEL);
    k_cast_bf16<<<gs(XP_COLS * D_INNER, TPB), TPB, 0, stream>>>(x_proj_w, wxp_bf, XP_COLS * D_INNER);
    k_cast_bf16<<<gs(D_INNER * DT_RANK, TPB), TPB, 0, stream>>>(dt_proj_w, wdt_bf, D_INNER * DT_RANK);
    k_cast_bf16<<<gs(D_MODEL * D_INNER, TPB), TPB, 0, stream>>>(out_proj_w, wout_bf, D_MODEL * D_INNER);

    // 2) in_proj GEMM: [4096,1024] x [4096,1024]^T -> split x | z
    //    block tile 64x128, wave tile 32x32, BK=64
    k_wmma_gemm<2, 2, 2, 4, 64, 1><<<dim3(BL / 64, (2 * D_INNER) / 128), TPB, 0, stream>>>(
        hs_bf, D_MODEL, win_bf, D_MODEL, D_MODEL,
        xf, zf, D_INNER, D_INNER, nullptr, nullptr);

    // 3) causal depthwise conv + SiLU
    k_conv_silu<<<gs(BL * D_INNER, TPB), TPB, 0, stream>>>(xf, conv_w, conv_b, xcf, xc_bf);

    // 4) x_proj GEMM: [4096,2048] x [96,2048]^T -> xp (f32 + bf16)
    //    block tile 64x96, wave tile 16x48
    k_wmma_gemm<1, 3, 4, 2, 64, 3><<<dim3(BL / 64, 1), TPB, 0, stream>>>(
        xc_bf, D_INNER, wxp_bf, D_INNER, D_INNER,
        xpf, nullptr, XP_COLS, 0, nullptr, xp_bf);

    // 5) dt GEMM: xp[:, :64] x [2048,64]^T + bias -> softplus -> dt  (K=64, 1 stage)
    k_wmma_gemm<2, 2, 2, 4, 64, 2><<<dim3(BL / 64, D_INNER / 128), TPB, 0, stream>>>(
        xp_bf, XP_COLS, wdt_bf, DT_RANK, DT_RANK,
        dtf, nullptr, D_INNER, 0, dt_proj_b, nullptr);

    // 6) selective scan + D skip + silu(z) gate -> y (bf16)
    k_scan<<<dim3(D_INNER / 16, NBATCH), TPB, 0, stream>>>(
        dtf, xcf, xpf, zf, A_log, Dvec, y_bf);

    // 7) out_proj GEMM: [4096,2048] x [1024,2048]^T -> d_out
    k_wmma_gemm<2, 2, 2, 4, 64, 0><<<dim3(BL / 64, D_MODEL / 128), TPB, 0, stream>>>(
        y_bf, D_INNER, wout_bf, D_INNER, D_INNER,
        out, nullptr, D_MODEL, 0, nullptr, nullptr);

    (void)in_sizes; (void)n_in; (void)out_size; (void)ws_size;
}
